// MotionCompensation_31568009626204
// MI455X (gfx1250) — matrix operand order repaired and checked by reading, not compiled, not run on hardware
//
#include <hip/hip_runtime.h>

// ---------------------------------------------------------------------------
// MotionCompensation pipeline for MI455X (gfx1250, wave32, WMMA 16x16x32 bf16)
//
// B=2, C=64, H=W=192, KS=3, N=9, L=3 resblocks.
// All convs are implicit GEMMs on v_wmma_f32_16x16x32_bf16 (f32 accumulate).
// Weight tiles are staged global->LDS with global_load_async_to_lds_b128
// (ASYNCcnt path), overlapped with the float->bf16 input-patch staging.
// ---------------------------------------------------------------------------

typedef __attribute__((ext_vector_type(16))) __bf16 v16bf;
typedef __attribute__((ext_vector_type(8)))  __bf16 v8bf;
typedef __attribute__((ext_vector_type(8)))  float  v8f;

static __device__ __forceinline__ __bf16 f2bf(float f) { return (__bf16)f; }

// LDS byte offset of a generic pointer into a __shared__ object.
static __device__ __forceinline__ unsigned lds_off(const void* p) {
  return (unsigned)(unsigned long long)(__attribute__((address_space(3))) const char*)p;
}

// Async DMA: 16B per lane, global -> LDS, tracked with ASYNCcnt.
static __device__ __forceinline__ void async_g2l_b128(unsigned lds_byte,
                                                      unsigned long long gaddr) {
  asm volatile("global_load_async_to_lds_b128 %0, %1, off"
               :: "v"(lds_byte), "v"(gaddr) : "memory");
}

static __device__ __forceinline__ void wait_async0() {
#if __has_builtin(__builtin_amdgcn_s_wait_asynccnt)
  __builtin_amdgcn_s_wait_asynccnt(0);
#else
  asm volatile("s_wait_asynccnt 0x0" ::: "memory");
#endif
}

// Build a 16x32 bf16 A-fragment from a row of 32 contiguous K values.
// Order used here (contiguous K, not AMD's table order): lane L: M = L%16, h = L/16;
//   element e (0..15): K = e + 16h (matches B fragment order)
// -> two contiguous 8-element (16B) loads at K offsets 16h and 16h+8.
static __device__ __forceinline__ v16bf load_a_frag(const __bf16* rowK, int hh) {
  union { v16bf v; v8bf h[2]; } u;
  u.h[0] = *(const v8bf*)(rowK + 16 * hh);
  u.h[1] = *(const v8bf*)(rowK + 16 * hh + 8);
  return u.v;
}

// B-fragment (32x16 bf16): lane L: N = L%16, h = L/16; element e: K = e + 16h
// -> one contiguous 16-element (32B) load when K (channel) is innermost.

// ---------------------------------------------------------------------------
// Weight prepack: (O, I, K, K) float OIHW -> chunk-major bf16:
//   dst[(chunk*taps + t)*Opad + o][32] with chunk = i/32, so each 32-channel
//   K-chunk is one contiguous (taps*Opad*64)-byte block whose layout is
//   byte-identical to the LDS tile [t][oc][32] -> eligible for flat async DMA.
// Serves normal convs, the deform GEMM (einsum 'ocn'), and GDN gamma (K=1).
// ---------------------------------------------------------------------------
__global__ void pack_kernel(const float* __restrict__ src, __bf16* __restrict__ dst,
                            int O, int Opad, int I, int K) {
  int idx = blockIdx.x * 256 + threadIdx.x;
  int taps = K * K;
  int total = Opad * taps * I;
  if (idx >= total) return;
  int cl = idx & 31;
  int rest = idx >> 5;
  int o = rest % Opad;
  int rest2 = rest / Opad;
  int t = rest2 % taps;
  int ch = rest2 / taps;
  int i = ch * 32 + cl;
  float v = 0.f;
  if (o < O) v = src[(((size_t)o * I + i) * K + t / K) * K + (t % K)];
  dst[idx] = f2bf(v);
}

// ---------------------------------------------------------------------------
// Generic implicit-GEMM conv (KSxKS, stride 1, pad `pad`), WMMA bf16.
// Block = 256 threads = 8 waves = NOCG oc-groups x (8/NOCG) pixel-groups.
// Each wave owns a 16(oc) x 16(px) f32 accumulator tile (v8f per lane).
// Input may be a channel-concat of two tensors (inA: c<CsplitA, inB: rest).
// Optional: square inputs while staging (GDN), bias add, residual add.
// ---------------------------------------------------------------------------
template <int KS, int NOCG>
__global__ __launch_bounds__(256)
void conv_wmma_kernel(const float* __restrict__ inA, const float* __restrict__ inB,
                      int CsplitA, int Cin, int Hin, int Win,
                      const __bf16* __restrict__ wpk, const float* __restrict__ bias,
                      int Cout, float* __restrict__ out, int Hout, int Wout,
                      int pad, const float* __restrict__ res, int inSquare) {
  constexpr int TAPS = KS * KS;
  constexpr int COP = NOCG * 16;        // padded Cout tile handled per block
  constexpr int PXG = 8 / NOCG;         // pixel groups
  constexpr int PW = PXG * 16;          // pixels per block (along W)
  constexpr int ROWS = (KS == 3) ? 3 : 1;
  constexpr int PATCHW = PW + (KS - 1);
  constexpr int WBYTES = TAPS * COP * 64;  // bf16 chunk bytes: taps*COP*32*2

  __shared__ __align__(16) __bf16 sW[TAPS * COP * 32];     // [t][oc][32c]
  __shared__ __align__(16) __bf16 sIn[ROWS * PATCHW * 32]; // [row][col][32c]

  const int tid = threadIdx.x, lane = tid & 31, wid = tid >> 5;
  const int ocg = wid % NOCG, pxg = wid / NOCG;
  const int x0 = blockIdx.x * PW, y = blockIdx.y, b = blockIdx.z;
  const int ln = lane & 15, hh = lane >> 4;
  const int CinB = Cin - CsplitA;

  v8f acc = {};
  __builtin_prefetch(wpk, 0, 1);  // global_prefetch_b8 on L2-resident weights

  const unsigned sW_base = lds_off(sW);

  for (int c0 = 0; c0 < Cin; c0 += 32) {
    // --- kick off async DMA of this chunk's weight tile (no VGPR roundtrip) ---
    {
      unsigned long long gsrc =
          (unsigned long long)(const char*)(wpk) + (unsigned long long)(c0 >> 5) * WBYTES;
      for (int i = tid * 16; i < WBYTES; i += 4096)  // 256 lanes * 16B
        async_g2l_b128(sW_base + i, gsrc + i);
    }
    // --- stage input patch (channel-contiguous), float->bf16 on the fly,
    //     overlapping with the async weight DMA ---
    {
      constexpr int NEL = ROWS * PATCHW * 32;
      for (int i = tid; i < NEL; i += 256) {
        int cl = i / (ROWS * PATCHW);
        int pos = i % (ROWS * PATCHW);
        int row = pos / PATCHW, col = pos % PATCHW;
        int gy = y - pad + row;
        int gx = x0 - pad + col;
        int c = c0 + cl;
        float v = 0.f;
        if (gy >= 0 && gy < Hin && gx >= 0 && gx < Win) {
          if (c < CsplitA)
            v = inA[(((size_t)b * CsplitA + c) * Hin + gy) * Win + gx];
          else
            v = inB[(((size_t)b * CinB + (c - CsplitA)) * Hin + gy) * Win + gx];
          if (inSquare) v *= v;
        }
        sIn[pos * 32 + cl] = f2bf(v);
      }
    }
    wait_async0();     // this wave's async LDS writes have landed
    __syncthreads();   // all waves' staging visible

#pragma unroll
    for (int t = 0; t < TAPS; ++t) {
      const int ky = t / KS, kx = t % KS;
      v16bf a = load_a_frag(&sW[(t * COP + (ocg * 16 + ln)) * 32], hh);
      const int px = pxg * 16 + ln;
      v16bf bf = *(const v16bf*)&sIn[((ky * PATCHW) + px + kx) * 32 + 16 * hh];
      acc = __builtin_amdgcn_wmma_f32_16x16x32_bf16(false, a, false, bf,
                                                    (short)0, acc, false, false);
    }
    __syncthreads();
  }

  // --- store: D layout lane L: N=L%16, VGPR j -> M = j + 8*(L/16) ---
  const int px = x0 + pxg * 16 + ln;
  if (px < Wout) {
#pragma unroll
    for (int j = 0; j < 8; ++j) {
      int oc = ocg * 16 + j + 8 * hh;
      if (oc < Cout) {
        size_t idx = (((size_t)b * Cout + oc) * Hout + y) * Wout + px;
        float v = acc[j];
        if (bias) v += bias[oc];
        if (res) v += res[idx];
        out[idx] = v;
      }
    }
  }
}

// ---------------------------------------------------------------------------
// Deformable conv (DCNv1): bilinear gather -> per-pixel GEMM over (n=9, c=64).
// Block = 32 pixels of one row; 8 waves = 4 ocg x 2 pxg. A streamed from
// global (chunk-major packed bf16, L2-resident); B tile gathered into LDS.
// ---------------------------------------------------------------------------
__global__ __launch_bounds__(256)
void deform_wmma_kernel(const float* __restrict__ fr, const float* __restrict__ off,
                        const __bf16* __restrict__ wdcp, float* __restrict__ xout,
                        int C, int H, int W) {
  __shared__ __align__(16) __bf16 sB[9 * 32 * 64]; // [n][px][c]
  __shared__ float sG[9 * 32 * 4];                 // bilinear weights
  __shared__ short sQ[9 * 32 * 4];                 // integer corners

  const int tid = threadIdx.x, lane = tid & 31, wid = tid >> 5;
  const int ocg = wid & 3, pxg = wid >> 2;
  const int x0 = blockIdx.x * 32, y = blockIdx.y, b = blockIdx.z;
  const int Hp = H + 2, Wp = W + 2;  // pad=1 on each side
  const int ln = lane & 15, hh = lane >> 4;

  // --- sampling coordinates + bilinear weights (matches reference exactly) ---
  for (int i = tid; i < 9 * 32; i += 256) {
    int n = i / 32, px = i % 32;
    int x = x0 + px;
    float knr = (float)(n / 3 - 1), knc = (float)(n % 3 - 1);
    size_t obase = (size_t)b * 18 * H * W;
    float orow = off[obase + ((size_t)n * H + y) * W + x];
    float ocol = off[obase + ((size_t)(9 + n) * H + y) * W + x];
    float pr_raw = (float)(y + 1) + knr + orow;
    float pc_raw = (float)(x + 1) + knc + ocol;
    float pr = fminf(fmaxf(pr_raw, 0.f), (float)(Hp - 1));
    float pc = fminf(fmaxf(pc_raw, 0.f), (float)(Wp - 1));
    float fr0 = floorf(pr_raw), fc0 = floorf(pc_raw);
    int qr0 = min(max((int)fr0, 0), Hp - 1);
    int qc0 = min(max((int)fc0, 0), Wp - 1);
    int qr1 = min(max((int)fr0 + 1, 0), Hp - 1);
    int qc1 = min(max((int)fc0 + 1, 0), Wp - 1);
    sG[i * 4 + 0] = (1.f + ((float)qr0 - pr)) * (1.f + ((float)qc0 - pc)); // lt
    sG[i * 4 + 1] = (1.f - ((float)qr1 - pr)) * (1.f - ((float)qc1 - pc)); // rb
    sG[i * 4 + 2] = (1.f + ((float)qr0 - pr)) * (1.f - ((float)qc1 - pc)); // lb
    sG[i * 4 + 3] = (1.f - ((float)qr1 - pr)) * (1.f + ((float)qc0 - pc)); // rt
    sQ[i * 4 + 0] = (short)qr0; sQ[i * 4 + 1] = (short)qc0;
    sQ[i * 4 + 2] = (short)qr1; sQ[i * 4 + 3] = (short)qc1;
  }
  __syncthreads();

  // --- gather bilinear samples into the B tile (bf16, channel-contiguous) ---
  for (int i = tid; i < 9 * 32 * 64; i += 256) {
    int c = i & 63;
    int pn = i >> 6;  // n*32 + px
    float glt = sG[pn * 4 + 0], grb = sG[pn * 4 + 1];
    float glb = sG[pn * 4 + 2], grt = sG[pn * 4 + 3];
    int qr0 = sQ[pn * 4 + 0], qc0 = sQ[pn * 4 + 1];
    int qr1 = sQ[pn * 4 + 2], qc1 = sQ[pn * 4 + 3];
    const float* base = fr + ((size_t)b * C + c) * H * W;
    auto X = [&](int r, int cc) -> float {
      r -= 1; cc -= 1;  // padded coords -> real image, zero outside
      return (r >= 0 && r < H && cc >= 0 && cc < W) ? base[(size_t)r * W + cc] : 0.f;
    };
    float v = glt * X(qr0, qc0) + grb * X(qr1, qc1) +
              glb * X(qr0, qc1) + grt * X(qr1, qc0);
    sB[pn * 64 + c] = f2bf(v);
  }
  __syncthreads();

  // --- GEMM: K = 9 taps x 64 channels = 18 chunks of 32, fully unrolled.
  //     Chunk-major weight layout: [(chunk*9 + n)*64 + oc][32] ---
  v8f acc = {};
#pragma unroll
  for (int ch = 0; ch < 2; ++ch) {      // c0 = ch*32
#pragma unroll
    for (int n = 0; n < 9; ++n) {
      const __bf16* arow = wdcp + (((size_t)ch * 9 + n) * 64 + (ocg * 16 + ln)) * 32;
      v16bf a = load_a_frag(arow, hh);
      int px = pxg * 16 + ln;
      v16bf bf = *(const v16bf*)&sB[(n * 32 + px) * 64 + ch * 32 + 16 * hh];
      acc = __builtin_amdgcn_wmma_f32_16x16x32_bf16(false, a, false, bf,
                                                    (short)0, acc, false, false);
    }
  }

  int px = x0 + pxg * 16 + ln;
#pragma unroll
  for (int j = 0; j < 8; ++j) {
    int oc = ocg * 16 + j + 8 * hh;
    xout[(((size_t)b * C + oc) * H + y) * W + px] = acc[j];
  }
}

// ---------------------------------------------------------------------------
// Elementwise: h = celu(h * rsqrt(norm))  (celu alpha=1: x>0 ? x : expm1(x))
// ---------------------------------------------------------------------------
__global__ void gdn_celu_kernel(float* __restrict__ h, const float* __restrict__ nrm,
                                size_t n) {
  size_t i = (size_t)blockIdx.x * 256 + threadIdx.x;
  if (i < n) {
    float v = h[i] * rsqrtf(nrm[i]);
    h[i] = v > 0.f ? v : expm1f(v);
  }
}

__global__ void add_kernel(const float* __restrict__ a, const float* __restrict__ b,
                           float* __restrict__ o, size_t n) {
  size_t i = (size_t)blockIdx.x * 256 + threadIdx.x;
  if (i < n) o[i] = a[i] + b[i];
}

// ---------------------------------------------------------------------------
extern "C" void kernel_launch(void* const* d_in, const int* in_sizes, int n_in,
                              void* d_out, int out_size, void* d_ws, size_t ws_size,
                              hipStream_t stream) {
  const float* f_r      = (const float*)d_in[0];
  const float* m_t      = (const float*)d_in[1];
  const float* w_pconv  = (const float*)d_in[2];
  const float* b_pconv  = (const float*)d_in[3];
  const float* w_dc     = (const float*)d_in[4];
  const float* w_cat    = (const float*)d_in[5];
  const float* b_cat    = (const float*)d_in[6];
  const float* rb_w1    = (const float*)d_in[7];
  const float* rb_w2    = (const float*)d_in[8];
  const float* rb_beta  = (const float*)d_in[9];
  const float* rb_gamma = (const float*)d_in[10];
  float* out = (float*)d_out;

  constexpr int Bb = 2, C = 64, H = 192, W = 192;
  constexpr int Hh = 190, Wh = 190;
  const size_t HW = (size_t)H * W;
  const size_t HWh = (size_t)Hh * Wh;

  // Workspace carve-out (deterministic, fully initialized before use)
  char* p = (char*)d_ws;
  auto carve = [&](size_t bytes) -> char* {
    char* r = p;
    p += (bytes + 255) & ~((size_t)255);
    return r;
  };
  float* offBuf = (float*)carve(sizeof(float) * Bb * 18 * HW);
  float* xBuf   = (float*)carve(sizeof(float) * Bb * C * HW);
  float* yA     = (float*)carve(sizeof(float) * Bb * C * HW);
  float* yB     = (float*)carve(sizeof(float) * Bb * C * HW);
  float* hBuf   = (float*)carve(sizeof(float) * Bb * C * HWh);
  float* nBuf   = (float*)carve(sizeof(float) * Bb * C * HWh);
  __bf16* Ppconv = (__bf16*)carve(2 * 32 * 9 * 64);
  __bf16* Pdc    = (__bf16*)carve(2 * 64 * 9 * 64);
  __bf16* Pcat   = (__bf16*)carve(2 * 64 * 9 * 128);
  __bf16* Prb1   = (__bf16*)carve(2 * 3 * 64 * 9 * 64);
  __bf16* Prb2   = (__bf16*)carve(2 * 3 * 64 * 9 * 64);
  __bf16* Pgdn   = (__bf16*)carve(2 * 3 * 64 * 64);

  auto pack = [&](const float* src, __bf16* dst, int O, int Opad, int I, int K) {
    int total = Opad * K * K * I;
    pack_kernel<<<dim3((total + 255) / 256), dim3(256), 0, stream>>>(src, dst, O, Opad, I, K);
  };
  pack(w_pconv, Ppconv, 18, 32, 64, 3);
  pack(w_dc, Pdc, 64, 64, 64, 3);
  pack(w_cat, Pcat, 64, 64, 128, 3);
  for (int i = 0; i < 3; ++i) {
    pack(rb_w1 + (size_t)i * 36864, Prb1 + (size_t)i * 36864, 64, 64, 64, 3);
    pack(rb_w2 + (size_t)i * 36864, Prb2 + (size_t)i * 36864, 64, 64, 64, 3);
    pack(rb_gamma + (size_t)i * 4096, Pgdn + (size_t)i * 4096, 64, 64, 64, 1);
  }

  auto conv = [&](const float* inA, const float* inB, int CsplitA, int Cin,
                  int Hin, int Win, const __bf16* wpk, const float* bias,
                  int Cout, int CoutPad, int ks, float* o, int Hout, int Wout,
                  int pad, const float* res, int sq) {
    if (ks == 3 && CoutPad == 64) {
      dim3 g((Wout + 31) / 32, Hout, Bb);
      conv_wmma_kernel<3, 4><<<g, 256, 0, stream>>>(inA, inB, CsplitA, Cin, Hin, Win,
                                                    wpk, bias, Cout, o, Hout, Wout, pad, res, sq);
    } else if (ks == 3) {
      dim3 g((Wout + 63) / 64, Hout, Bb);
      conv_wmma_kernel<3, 2><<<g, 256, 0, stream>>>(inA, inB, CsplitA, Cin, Hin, Win,
                                                    wpk, bias, Cout, o, Hout, Wout, pad, res, sq);
    } else {
      dim3 g((Wout + 31) / 32, Hout, Bb);
      conv_wmma_kernel<1, 4><<<g, 256, 0, stream>>>(inA, inB, CsplitA, Cin, Hin, Win,
                                                    wpk, bias, Cout, o, Hout, Wout, pad, res, sq);
    }
  };

  // 1) offset = conv3x3(m_t, w_pconv, b_pconv, pad=1): 64 -> 18 (padded to 32)
  conv(m_t, nullptr, 64, 64, H, W, Ppconv, b_pconv, 18, 32, 3, offBuf, H, W, 1, nullptr, 0);

  // 2) x = deform_conv(f_r, offset, w_dc)
  deform_wmma_kernel<<<dim3(W / 32, H, Bb), 256, 0, stream>>>(f_r, offBuf, Pdc, xBuf, C, H, W);

  // 3) y = conv3x3(concat(x, f_r), w_cat, b_cat, pad=1): 128 -> 64
  conv(xBuf, f_r, 64, 128, H, W, Pcat, b_cat, 64, 64, 3, yA, H, W, 1, nullptr, 0);

  // 4) ResBlocks
  float* cur = yA;
  float* nxt = yB;
  for (int i = 0; i < 3; ++i) {
    // conv1 (no pad): 192 -> 190
    conv(cur, nullptr, 64, 64, H, W, Prb1 + (size_t)i * 36864, nullptr,
         64, 64, 3, hBuf, Hh, Wh, 0, nullptr, 0);
    // GDN norm = gamma @ h^2 + beta (1x1 GEMM, square while staging)
    conv(hBuf, nullptr, 64, 64, Hh, Wh, Pgdn + (size_t)i * 4096, rb_beta + (size_t)i * 64,
         64, 64, 1, nBuf, Hh, Wh, 0, nullptr, 1);
    // h = celu(h * rsqrt(norm))
    size_t nh = (size_t)Bb * C * HWh;
    gdn_celu_kernel<<<dim3((unsigned)((nh + 255) / 256)), 256, 0, stream>>>(hBuf, nBuf, nh);
    // conv2 (pad=2): 190 -> 192, fused residual add
    conv(hBuf, nullptr, 64, 64, Hh, Wh, Prb2 + (size_t)i * 36864, nullptr,
         64, 64, 3, nxt, H, W, 2, cur, 0);
    float* t = cur; cur = nxt; nxt = t;
  }

  // 5) out = y + x
  size_t nt = (size_t)Bb * C * HW;
  add_kernel<<<dim3((unsigned)((nt + 255) / 256)), 256, 0, stream>>>(cur, xBuf, out, nt);
}
